// PTFDecoder_66992899883085
// MI455X (gfx1250) — matrix operand order repaired and checked by reading, not compiled, not run on hardware
//
#include <hip/hip_runtime.h>

// ---------------- problem constants ----------------
constexpr int Jn = 14;
constexpr int Dn = 3;
constexpr int Cn = 128;
constexpr int Hn = 128;
constexpr int Bn = 2;
constexpr int Tn = 32768;
constexpr int TT = 128;        // t-tile per block
constexpr int KP = 136;        // LDS channel stride (272B rows -> 16B aligned chunks)

// ---------------- types ----------------
typedef __bf16 bf16_t;
typedef bf16_t v16bf __attribute__((ext_vector_type(16)));
typedef float  v8f   __attribute__((ext_vector_type(8)));

struct __attribute__((aligned(16))) U4 { unsigned int x, y, z, w; };
struct __attribute__((aligned(16))) F4 { float x, y, z, w; };

union Frag16 { U4 q[2]; v16bf v; };

__device__ __forceinline__ unsigned short f2bf(float f) {
    unsigned int u = __float_as_uint(f);
    unsigned int r = u + 0x7FFFu + ((u >> 16) & 1u);   // round-to-nearest-even
    return (unsigned short)(r >> 16);
}
__device__ __forceinline__ float bflo(unsigned int u) { return __uint_as_float(u << 16); }
__device__ __forceinline__ float bfhi(unsigned int u) { return __uint_as_float(u & 0xFFFF0000u); }

// ---------------- shared memory ----------------
__shared__ __align__(16) unsigned short sCbuf[Cn * KP];   // c tile, [t][ch] bf16 (persistent)
__shared__ __align__(16) unsigned short sH1[Hn * KP];     // activations ping
__shared__ __align__(16) unsigned short sH2[Hn * KP];     // activations pong
__shared__ __align__(16) unsigned short sW[Hn * KP];      // current layer weights bf16 [m][k]
__shared__ float sBias[Hn];
__shared__ float sW0p[Dn * Hn];                            // [d][m] fp32, layer-0 p-columns
__shared__ float sW3[Hn];

// A fragment: 16xK slab of weights for this wave's rows at K-step kk.
__device__ __forceinline__ void load_a(Frag16& a, int mA, int koffA, int kk) {
    const unsigned short* ap = sW + mA * KP + kk * 32 + koffA;
    a.q[0] = *(const U4*)(ap);        // K = k0+koffA .. +7
    a.q[1] = *(const U4*)(ap + 16);   // K = k0+16+koffA .. +7
}
// B fragment: Kx16 slab of activations for N-tile n at K-step kk.
__device__ __forceinline__ void load_b(Frag16& b, const unsigned short* __restrict__ actIn,
                                       int lm, int koffB, int kk, int n) {
    const unsigned short* bp = actIn + (n * 16 + lm) * KP + kk * 32 + koffB;
    b.q[0] = *(const U4*)(bp);        // K = k0+koffB .. +7
    b.q[1] = *(const U4*)(bp + 16);   // K = k0+16+koffB .. +7
}

// One GEMM layer: out[m][t] = relu(W[m][:]*in[:][t] + bias[m] (+ W0p*p)).
// Fragment loads are software-pipelined one WMMA ahead (double-buffered A/B)
// so the ds_load latency overlaps with matrix-pipe execution instead of
// serializing on s_wait_dscnt 0 before every WMMA.
template <bool ADDP>
__device__ __forceinline__ void gemm_layer(const unsigned short* __restrict__ actIn,
                                           unsigned short* __restrict__ actOut,
                                           const float* __restrict__ pG,  // &p[b][j*3+0][t0]
                                           int lane, int mbase)
{
    const int lm    = lane & 15;
    const int hi    = lane >> 4;
    const int mA    = mbase + lm;
    const int koffA = hi * 8;    // 16-bit A layout: lanes 16-31 hold K+8 pairs
    const int koffB = hi * 8;    // 16-bit B layout: lanes 16-31 hold K+8 pairs

    v8f acc[8];
    #pragma unroll
    for (int n = 0; n < 8; ++n) {
        #pragma unroll
        for (int g = 0; g < 8; ++g) acc[n][g] = 0.0f;
    }

    Frag16 a[2], b[2];
    load_a(a[0], mA, koffA, 0);
    load_b(b[0], actIn, lm, koffB, 0, 0);

    #pragma unroll
    for (int kk = 0; kk < 4; ++kk) {
        #pragma unroll
        for (int n = 0; n < 8; ++n) {
            const int cur = (kk * 8 + n) & 1;
            // prefetch the next fragment(s) before consuming the current ones
            if (n == 7) {
                if (kk < 3) {
                    load_a(a[(kk + 1) & 1], mA, koffA, kk + 1);
                    load_b(b[cur ^ 1], actIn, lm, koffB, kk + 1, 0);
                }
            } else {
                load_b(b[cur ^ 1], actIn, lm, koffB, kk, n + 1);
            }
            acc[n] = __builtin_amdgcn_wmma_f32_16x16x32_bf16(
                         false, a[kk & 1].v, false, b[cur].v, (short)0, acc[n], false, false);
        }
    }

    // epilogue: bias (+p-channel correction), relu, pack bf16 pairs into LDS [t][ch]
    #pragma unroll
    for (int n = 0; n < 8; ++n) {
        const int tl = n * 16 + lm;
        float p0 = 0.f, p1 = 0.f, p2 = 0.f;
        if (ADDP) {
            p0 = pG[tl];
            p1 = pG[(long)Tn + tl];
            p2 = pG[2L * Tn + tl];
        }
        #pragma unroll
        for (int g = 0; g < 8; g += 2) {
            const int m0 = mbase + hi * 8 + g;
            float v0 = acc[n][g]     + sBias[m0];
            float v1 = acc[n][g + 1] + sBias[m0 + 1];
            if (ADDP) {
                v0 += sW0p[m0] * p0 + sW0p[Hn + m0] * p1 + sW0p[2 * Hn + m0] * p2;
                v1 += sW0p[m0 + 1] * p0 + sW0p[Hn + m0 + 1] * p1 + sW0p[2 * Hn + m0 + 1] * p2;
            }
            v0 = fmaxf(v0, 0.0f);
            v1 = fmaxf(v1, 0.0f);
            unsigned int pk = (unsigned int)f2bf(v0) | ((unsigned int)f2bf(v1) << 16);
            *(unsigned int*)(actOut + tl * KP + m0) = pk;   // ds_store_b32
        }
    }
}

__device__ __forceinline__ void load_w_dense(const float* __restrict__ Wg,
                                             const float* __restrict__ bg, int tid)
{
    // 128x128 fp32 -> bf16 [m][k] via float4 loads + ds_store_b64
    #pragma unroll 4
    for (int it = 0; it < 16; ++it) {
        int idx = it * 256 + tid;
        int m = idx >> 5;
        int k = (idx & 31) * 4;
        F4 f = *(const F4*)(Wg + m * Hn + k);
        unsigned long long pk =
            (unsigned long long)((unsigned int)f2bf(f.x) | ((unsigned int)f2bf(f.y) << 16)) |
            ((unsigned long long)((unsigned int)f2bf(f.z) | ((unsigned int)f2bf(f.w) << 16)) << 32);
        *(unsigned long long*)(sW + m * KP + k) = pk;
    }
    if (tid < Hn) sBias[tid] = bg[tid];
}

__global__ __launch_bounds__(256) void ptf_decoder_kernel(
    const float* __restrict__ p,  const float* __restrict__ c,
    const float* __restrict__ ps,
    const float* __restrict__ W0, const float* __restrict__ b0,
    const float* __restrict__ W1, const float* __restrict__ b1,
    const float* __restrict__ W2, const float* __restrict__ b2,
    const float* __restrict__ W3, const float* __restrict__ b3,
    float* __restrict__ out)
{
    const int tid   = threadIdx.x;
    const int lane  = tid & 31;
    const int wave  = tid >> 5;
    const int mbase = wave * 16;
    const int tiles = Tn / TT;
    const int b     = blockIdx.x / tiles;
    const int t0    = (blockIdx.x % tiles) * TT;

    // ---- stage c tile: [ch][t] fp32 global -> [t][ch] bf16 LDS (once per block) ----
    {
        const float* cb = c + (size_t)b * Cn * Tn + t0;
        #pragma unroll 4
        for (int it = 0; it < 16; ++it) {
            int idx = it * 256 + tid;       // 4096 float4 chunks
            int ch  = idx >> 5;
            int t   = (idx & 31) * 4;
            F4 f = *(const F4*)(cb + (size_t)ch * Tn + t);
            sCbuf[(t + 0) * KP + ch] = f2bf(f.x);
            sCbuf[(t + 1) * KP + ch] = f2bf(f.y);
            sCbuf[(t + 2) * KP + ch] = f2bf(f.z);
            sCbuf[(t + 3) * KP + ch] = f2bf(f.w);
        }
    }
    __syncthreads();

    float oacc = 0.0f;   // threads 0..127 accumulate logits over j for t = t0 + tid

    for (int j = 0; j < Jn; ++j) {
        // ---- load W0 (c-columns), b0, W0 p-columns ----
        {
            const float* W0j = W0 + (size_t)j * Hn * (Cn + Dn);
            #pragma unroll 4
            for (int it = 0; it < 64; ++it) {
                int idx = it * 256 + tid;
                int m = idx >> 7, k = idx & 127;
                sW[m * KP + k] = f2bf(W0j[m * (Cn + Dn) + Dn + k]);
            }
            for (int idx = tid; idx < Dn * Hn; idx += 256) {
                int d = idx >> 7, m = idx & 127;
                sW0p[d * Hn + m] = W0j[m * (Cn + Dn) + d];
            }
            if (tid < Hn) sBias[tid] = b0[j * Hn + tid];
            if (tid == 0) __builtin_prefetch(W1 + (size_t)j * Hn * Hn, 0, 1);
        }
        __syncthreads();

        // ---- layer 0: Cbuf -> H1 (adds p contribution) ----
        gemm_layer<true>(sCbuf, sH1, p + ((size_t)b * Jn + j) * Dn * Tn + t0, lane, mbase);
        __syncthreads();

        load_w_dense(W1 + (size_t)j * Hn * Hn, b1 + j * Hn, tid);
        if (tid == 0) __builtin_prefetch(W2 + (size_t)j * Hn * Hn, 0, 1);
        __syncthreads();

        // ---- layer 1: H1 -> H2 ----
        gemm_layer<false>(sH1, sH2, nullptr, lane, mbase);
        __syncthreads();

        load_w_dense(W2 + (size_t)j * Hn * Hn, b2 + j * Hn, tid);
        if (tid < Hn) sW3[tid] = W3[j * Hn + tid];
        __syncthreads();

        // ---- layer 2: H2 -> H1 ----
        gemm_layer<false>(sH2, sH1, nullptr, lane, mbase);
        __syncthreads();

        // ---- layer 3: o[t] = W3 . h2[:,t] + b3 ; accumulate masked mean ----
        if (tid < TT) {
            const unsigned short* row = sH1 + tid * KP;  // contiguous 128 bf16
            float dot = 0.0f;
            #pragma unroll
            for (int cix = 0; cix < 16; ++cix) {
                U4 qv = *(const U4*)(row + cix * 8);      // ds_load_b128
                unsigned int us[4] = {qv.x, qv.y, qv.z, qv.w};
                #pragma unroll
                for (int e = 0; e < 4; ++e) {
                    int k = cix * 8 + e * 2;
                    dot += bflo(us[e]) * sW3[k] + bfhi(us[e]) * sW3[k + 1];
                }
            }
            float o   = dot + b3[j];
            float psv = ps[((size_t)b * Jn + j) * Tn + t0 + tid];
            oacc += o * psv;
        }
        __syncthreads();
    }

    if (tid < TT) out[(size_t)b * Tn + t0 + tid] = oacc * (1.0f / (float)Jn);
}

extern "C" void kernel_launch(void* const* d_in, const int* in_sizes, int n_in,
                              void* d_out, int out_size, void* d_ws, size_t ws_size,
                              hipStream_t stream) {
    (void)in_sizes; (void)n_in; (void)out_size; (void)d_ws; (void)ws_size;
    const float* p  = (const float*)d_in[0];
    // d_in[1] = z : unused by the reference computation
    const float* c  = (const float*)d_in[2];
    const float* ps = (const float*)d_in[3];
    const float* W0 = (const float*)d_in[4];
    const float* b0 = (const float*)d_in[5];
    const float* W1 = (const float*)d_in[6];
    const float* b1 = (const float*)d_in[7];
    const float* W2 = (const float*)d_in[8];
    const float* b2 = (const float*)d_in[9];
    const float* W3 = (const float*)d_in[10];
    const float* b3 = (const float*)d_in[11];
    float* out = (float*)d_out;

    dim3 grid(Bn * (Tn / TT));   // 512 blocks
    dim3 block(256);             // 8 wave32s
    ptf_decoder_kernel<<<grid, block, 0, stream>>>(p, c, ps, W0, b0, W1, b1,
                                                   W2, b2, W3, b3, out);
}
